// ExpandCoords_39737037423037
// MI455X (gfx1250) — matrix operand order repaired and checked by reading, not compiled, not run on hardware
//
#include <hip/hip_runtime.h>

// MI455X / gfx1250 implementation of the conv-BN-residual graph.
// All convs run as chains of v_wmma_f32_16x16x32_bf16:
//   M = 16 output channels, N = 16 voxels (one W-row), K = 2 taps x 16 in-ch.
// BN scales folded into bf16 weights; BN shifts added in the epilogue.
// Stage geometry is template-constant: tap offsets fold to DS-immediates.
// Each wave drives 8 independent accumulators (8 output d-rows); per tap-pair
// one base register (cndmask+add) serves all 16 B ds_loads via immediate row
// offsets, staged ahead of the 8 WMMAs so the DS counter drains against the
// matrix pipe.

typedef __bf16 v8bf  __attribute__((ext_vector_type(8)));
typedef __bf16 v16bf __attribute__((ext_vector_type(16)));
typedef float  v8f   __attribute__((ext_vector_type(8)));

#define DD 128
#define HH 128
#define WW 16
#define NVOX (DD * HH * WW)     // 262144
#define BD 8                    // d-rows per workgroup (one per accumulator)
#define BH 8                    // h-rows per workgroup (one per wave)
#define TDc (BD + 6)            // LDS tile extent in d (halo 3)
#define THc (BH + 6)            // LDS tile extent in h (halo 3)
#define TWc (WW + 4)            // LDS tile extent in w (halo 2)
#define SACT_USH (TDc * THc * TWc * 16)   // ushorts of activation tile
#define NTHREADS 256
#define NWAVES 8
#define EPSF 1e-5f

__device__ __forceinline__ unsigned short f2bf(float f) {
  unsigned int u = __float_as_uint(f);
  u += 0x7FFFu + ((u >> 16) & 1u);      // round-to-nearest-even
  return (unsigned short)(u >> 16);
}

__device__ __forceinline__ uint4 pack8(const unsigned short* u) {
  uint4 q;
  q.x = (unsigned)u[0] | ((unsigned)u[1] << 16);
  q.y = (unsigned)u[2] | ((unsigned)u[3] << 16);
  q.z = (unsigned)u[4] | ((unsigned)u[5] << 16);
  q.w = (unsigned)u[6] | ((unsigned)u[7] << 16);
  return q;
}

// Flat tap index -> byte offset inside the LDS activation tile (relative to
// the corner-biased per-row base). Constexpr so it folds to a literal inside
// the fully unrolled pair loop. Walk must match prep_w exactly.
__device__ __forceinline__ constexpr int tap_off_b(
    int t, int nconvs,
    int kd0, int kh0, int kw0, int kd1, int kh1, int kw1,
    int kd2, int kh2, int kw2, int nreal) {
  if (t >= nreal) return ((3 * THc + 3) * TWc + 2) * 32;  // pad tap: center
  int KD = kd0, KH = kh0, KW = kw0, rem = t;
  if (nconvs > 1 && rem >= kd0 * kh0 * kw0) {
    rem -= kd0 * kh0 * kw0; KD = kd1; KH = kh1; KW = kw1;
    if (nconvs > 2 && rem >= kd1 * kh1 * kw1) {
      rem -= kd1 * kh1 * kw1; KD = kd2; KH = kh2; KW = kw2;
    }
  }
  int kdi = rem / (KH * KW);
  int r = rem % (KH * KW);
  int khi = r / KW;
  int kwi = r % KW;
  int dzb = kdi - (KD >> 1) + 3;
  int dyb = khi - (KH >> 1) + 3;
  int dxb = kwi - (KW >> 1) + 2;
  return ((dzb * THc + dyb) * TWc + dxb) * 32;
}

// Runtime version of the same walk (used by prep_w for weight packing).
__device__ __forceinline__ void tap_decode(
    int t, int nConvs,
    int kd0, int kh0, int kw0, int kd1, int kh1, int kw1,
    int kd2, int kh2, int kw2,
    int& j, int& KD, int& KH, int& KW, int& kdi, int& khi, int& kwi) {
  int c0 = kd0 * kh0 * kw0;
  int c1 = kd1 * kh1 * kw1;
  j = 0; KD = kd0; KH = kh0; KW = kw0;
  int rem = t;
  if (nConvs > 1 && rem >= c0) {
    rem -= c0; j = 1; KD = kd1; KH = kh1; KW = kw1;
    if (nConvs > 2 && rem >= c1) {
      rem -= c1; j = 2; KD = kd2; KH = kh2; KW = kw2;
    }
  }
  kdi = rem / (KH * KW);
  int r = rem % (KH * KW);
  khi = r / KW;
  kwi = r % KW;
}

// ---------- x (NCDHW f32) -> xb ([D][H][W][C] bf16) ----------
__global__ void prep_x(const float* __restrict__ x, unsigned short* __restrict__ xb) {
  int p = blockIdx.x * blockDim.x + threadIdx.x;
  if (p >= NVOX) return;
  unsigned short u[16];
#pragma unroll
  for (int c = 0; c < 16; ++c) u[c] = f2bf(x[(size_t)c * NVOX + p]);
  uint4* dst = (uint4*)(xb + (size_t)p * 16);
  dst[0] = pack8(u);
  dst[1] = pack8(u + 8);
}

// ---------- pack BN-scaled bf16 weights into WMMA A-matrix lane layout ----
__global__ void prep_w(const float* __restrict__ w0, const float* __restrict__ w1p,
                       const float* __restrict__ w2p,
                       int nConvs,
                       int kd0, int kh0, int kw0, int kd1, int kh1, int kw1,
                       int kd2, int kh2, int kw2,
                       int r0, int r1, int r2,
                       const float* __restrict__ gamma, const float* __restrict__ var,
                       unsigned short* __restrict__ blob, int nReal, int nPad) {
  int g = blockIdx.x * blockDim.x + threadIdx.x;
  if (g >= nPad * 32) return;
  int t = g >> 5, L = g & 31;
  unsigned short o8[8] = {0, 0, 0, 0, 0, 0, 0, 0};
  if (t < nReal) {
    int j, KD, KH, KW, kdi, khi, kwi;
    tap_decode(t, nConvs, kd0, kh0, kw0, kd1, kh1, kw1, kd2, kh2, kw2,
               j, KD, KH, KW, kdi, khi, kwi);
    const float* wp = (j == 0) ? w0 : (j == 1) ? w1p : w2p;
    int rj = (j == 0) ? r0 : (j == 1) ? r1 : r2;
    int m = L & 15;
    int cb = (L < 16) ? 0 : 8;
    float scale = 1.0f;
    if (rj >= 0) scale = gamma[rj * 16 + m] * rsqrtf(var[rj * 16 + m] + EPSF);
#pragma unroll
    for (int jj = 0; jj < 8; ++jj) {
      int c = cb + jj;
      float v = wp[(((m * 16 + c) * KD + kdi) * KH + khi) * KW + kwi] * scale;
      o8[jj] = f2bf(v);
    }
  }
  ((uint4*)blob)[g] = pack8(o8);
}

// ---------- fused WMMA conv stage -----------------------------------------
// MODE 0: outF = acc+sh ; outB = bf16(acc+sh)                (x1 stage)
// MODE 1: outF = acc+sh ; outB = bf16(relu(acc+sh))          (s234 / t1)
// MODE 2: outB = bf16(relu(acc+sh+add0+add1))                (xs stage)
// MODE 3: outF = acc+sh                                      (yacc stage)
// MODE 4: NCDHW d_out = relu(acc+add0+add1)                  (final)
template <int MODE, int NCONVS,
          int KD0, int KH0, int KW0, int KD1, int KH1, int KW1,
          int KD2, int KH2, int KW2,
          int R0, int R1, int R2, int NREAL, int NPAD>
__global__ __launch_bounds__(NTHREADS, 1) void conv_stage(
    const unsigned short* __restrict__ in,    // bf16 [D][H][W][16]
    const unsigned short* __restrict__ blob,  // packed bf16 weights
    const float* __restrict__ gamma, const float* __restrict__ beta,
    const float* __restrict__ mean, const float* __restrict__ var,
    float* __restrict__ outF,
    unsigned short* __restrict__ outB,
    const float* __restrict__ add0,
    const float* __restrict__ add1) {
  extern __shared__ char smemRaw[];
  unsigned short* sAct = (unsigned short*)smemRaw;  // SACT_USH ushorts
  unsigned short* sW = sAct + SACT_USH;             // NPAD*256 ushorts

  const int tid = threadIdx.x;
  const int dB = (blockIdx.x & 15) * BD;   // 128/BD = 16 blocks in d
  const int hB = (blockIdx.x >> 4) * BH;   // 128/BH = 16 blocks in h

  // --- weights: global -> LDS (16B chunks) ---
  {
    const uint4* gsrc = (const uint4*)blob;
    uint4* gdst = (uint4*)sW;
    for (int i = tid; i < NPAD * 32; i += NTHREADS) gdst[i] = gsrc[i];
  }

  // --- activation tile + halo: global -> LDS, zero-padded at boundaries ---
  {
    const uint4 z4 = make_uint4(0u, 0u, 0u, 0u);
    for (int vx = tid; vx < TDc * THc * TWc; vx += NTHREADS) {
      int td = vx / (THc * TWc);
      int rem = vx % (THc * TWc);
      int th = rem / TWc;
      int tw = rem % TWc;
      int gd = dB + td - 3, gh = hB + th - 3, gw = tw - 2;
      uint4 lo = z4, hi = z4;
      if (gd >= 0 && gd < DD && gh >= 0 && gh < HH && gw >= 0 && gw < WW) {
        const uint4* src = (const uint4*)(in + (size_t)((gd * HH + gh) * WW + gw) * 16);
        lo = src[0];
        hi = src[1];
      }
      uint4* dst = (uint4*)(sAct + (size_t)vx * 16);
      dst[0] = lo;
      dst[1] = hi;
    }
  }
  __syncthreads();

  const int wave = tid >> 5;
  const int lane = tid & 31;
  const int n = lane & 15;        // voxel column (N) and W coordinate
  const int khalf = lane >> 4;    // which tap of the K=32 pair this lane feeds
  const int cbase = khalf * 8;    // C/D channels owned by this lane

  // This wave's 8 output rows: (d = dB + r, h = hB + wave), r = 0..7.
  // Row r's LDS base is row 0's base + r*THc*TWc*32 (fits DS imm offset).
  const char* bB0 = (const char*)sAct + (size_t)(wave * TWc + n) * 32;
  const char* aB = (const char*)sW + lane * 16;

  v8f acc[BD];
#pragma unroll
  for (int r = 0; r < BD; ++r)
    acc[r] = (v8f){0.f, 0.f, 0.f, 0.f, 0.f, 0.f, 0.f, 0.f};

#pragma unroll
  for (int pr = 0; pr < NPAD / 2; ++pr) {
    const int t0 = 2 * pr, t1 = t0 + 1;
    // A: this half's tap; elements 0..7 = ch 0..7, 8..15 = ch 8..15
    const int tA = khalf ? t1 : t0;
    const v8bf alo = *(const v8bf*)((const char*)sW + tA * 512 + n * 16);
    const v8bf ahi = *(const v8bf*)((const char*)sW + tA * 512 + (16 + n) * 16);
    const v16bf a = __builtin_shufflevector(alo, ahi, 0, 1, 2, 3, 4, 5, 6, 7,
                                            8, 9, 10, 11, 12, 13, 14, 15);
    // B: this lane's tap offset is a compile-time literal per half; one base
    // register per pair, all row loads use DS-immediate offsets.
    const int o0 = tap_off_b(t0, NCONVS, KD0, KH0, KW0, KD1, KH1, KW1,
                             KD2, KH2, KW2, NREAL);
    const int o1 = tap_off_b(t1, NCONVS, KD0, KH0, KW0, KD1, KH1, KW1,
                             KD2, KH2, KW2, NREAL);
    const char* bp = bB0 + (khalf ? o1 : o0);
    // Stage all 8 row-B operands first: 16 ds_load_b128 in flight, then the
    // 8 WMMAs drain the DS counter with staggered waits.
    v16bf b[BD];
#pragma unroll
    for (int r = 0; r < BD; ++r)
      b[r] = *(const v16bf*)(bp + r * (THc * TWc * 32));
#pragma unroll
    for (int r = 0; r < BD; ++r)
      acc[r] = __builtin_amdgcn_wmma_f32_16x16x32_bf16(false, a, false, b[r],
                                                       (short)0, acc[r],
                                                       false, false);
  }

  // BN shift for this lane's 8 output channels
  float sh[8];
#pragma unroll
  for (int jj = 0; jj < 8; ++jj) {
    float s = 0.0f;
    int c = cbase + jj;
    if (R0 >= 0) s += beta[R0 * 16 + c] - mean[R0 * 16 + c] * (gamma[R0 * 16 + c] * rsqrtf(var[R0 * 16 + c] + EPSF));
    if (R1 >= 0) s += beta[R1 * 16 + c] - mean[R1 * 16 + c] * (gamma[R1 * 16 + c] * rsqrtf(var[R1 * 16 + c] + EPSF));
    if (R2 >= 0) s += beta[R2 * 16 + c] - mean[R2 * 16 + c] * (gamma[R2 * 16 + c] * rsqrtf(var[R2 * 16 + c] + EPSF));
    sh[jj] = s;
  }

  // ---- epilogue: 8 rows ----
#pragma unroll
  for (int r = 0; r < BD; ++r) {
    const int d = dB + r, h = hB + wave;
    const int p = (d * HH + h) * WW + n;
    float v[8];
#pragma unroll
    for (int jj = 0; jj < 8; ++jj) v[jj] = acc[r][jj] + sh[jj];
    if (MODE == 2 || MODE == 4) {
      v8f a0 = *(const v8f*)(add0 + (size_t)p * 16 + cbase);
      v8f a1 = *(const v8f*)(add1 + (size_t)p * 16 + cbase);
#pragma unroll
      for (int jj = 0; jj < 8; ++jj) v[jj] += a0[jj] + a1[jj];
    }
    if (MODE == 0 || MODE == 1 || MODE == 3) {
      v8f o;
#pragma unroll
      for (int jj = 0; jj < 8; ++jj) o[jj] = v[jj];
      *(v8f*)(outF + (size_t)p * 16 + cbase) = o;
    }
    if (MODE == 0 || MODE == 1 || MODE == 2) {
      unsigned short u[8];
#pragma unroll
      for (int jj = 0; jj < 8; ++jj) {
        float t = (MODE == 0) ? v[jj] : fmaxf(v[jj], 0.0f);
        u[jj] = f2bf(t);
      }
      *(uint4*)(outB + (size_t)p * 16 + cbase) = pack8(u);
    }
    if (MODE == 4) {
#pragma unroll
      for (int jj = 0; jj < 8; ++jj)
        outF[(size_t)(cbase + jj) * NVOX + p] = fmaxf(v[jj], 0.0f);
    }
  }
}

extern "C" void kernel_launch(void* const* d_in, const int* in_sizes, int n_in,
                              void* d_out, int out_size, void* d_ws, size_t ws_size,
                              hipStream_t stream) {
  const float* x = (const float*)d_in[0];
  const float* w1 = (const float*)d_in[1];
  const float* w2 = (const float*)d_in[2];
  const float* w3 = (const float*)d_in[3];
  const float* w4 = (const float*)d_in[4];
  const float* w5 = (const float*)d_in[5];
  const float* w6 = (const float*)d_in[6];
  const float* w7 = (const float*)d_in[7];
  const float* wch = (const float*)d_in[8];
  const float* wr1 = (const float*)d_in[9];
  const float* wr2 = (const float*)d_in[10];
  const float* wr3 = (const float*)d_in[11];
  const float* bn_gamma = (const float*)d_in[12];
  const float* bn_beta = (const float*)d_in[13];
  const float* bn_mean = (const float*)d_in[14];
  const float* bn_var = (const float*)d_in[15];
  float* out = (float*)d_out;

  const size_t MB = 1024 * 1024;
  char* ws = (char*)d_ws;
  unsigned short* xb = (unsigned short*)(ws + 0);        // 8 MB bf16
  float* x1f = (float*)(ws + 8 * MB);                    // 16 MB f32
  unsigned short* x1b = (unsigned short*)(ws + 24 * MB); // 8 MB bf16
  float* s234 = (float*)(ws + 32 * MB);                  // 16 MB f32
  unsigned short* t1b = (unsigned short*)(ws + 48 * MB); // 8 MB bf16
  unsigned short* xsb = x1b;                             // reuse after stage 3
  float* yacc = s234;                                    // reuse after stage 4
  unsigned short* blob2 = (unsigned short*)(ws + 56 * MB);
  unsigned short* blob3 = (unsigned short*)(ws + 56 * MB + 256 * 1024);
  unsigned short* blob4 = (unsigned short*)(ws + 56 * MB + 512 * 1024);
  unsigned short* blob5a = (unsigned short*)(ws + 56 * MB + 768 * 1024);
  unsigned short* blob5b = (unsigned short*)(ws + 56 * MB + 1024 * 1024);

  // ---- prep ----
  prep_x<<<NVOX / NTHREADS, NTHREADS, 0, stream>>>(x, xb);

  auto wgrid = [](int nPad) { return (nPad * 32 + NTHREADS - 1) / NTHREADS; };
  prep_w<<<wgrid(10), NTHREADS, 0, stream>>>(w1, w1, w1, 1, 3, 3, 1, 1, 1, 1, 1, 1, 1,
                                             0, -1, -1, bn_gamma, bn_var, blob2, 9, 10);
  prep_w<<<wgrid(330), NTHREADS, 0, stream>>>(w2, w3, w4, 3, 3, 3, 1, 5, 5, 3, 7, 7, 5,
                                              1, 2, 3, bn_gamma, bn_var, blob3, 329, 330);
  prep_w<<<wgrid(330), NTHREADS, 0, stream>>>(w5, w6, w7, 3, 3, 3, 1, 5, 5, 3, 7, 7, 5,
                                              4, 5, 6, bn_gamma, bn_var, blob4, 329, 330);
  prep_w<<<wgrid(330), NTHREADS, 0, stream>>>(wr1, wr2, wr3, 3, 3, 3, 1, 5, 5, 3, 7, 7, 5,
                                              7, 8, 9, bn_gamma, bn_var, blob5a, 329, 330);
  prep_w<<<wgrid(2), NTHREADS, 0, stream>>>(wch, wch, wch, 1, 1, 1, 1, 1, 1, 1, 1, 1, 1,
                                            -1, -1, -1, bn_gamma, bn_var, blob5b, 1, 2);

  const int grid = (DD / BD) * (HH / BH);  // 256 workgroups
  auto shsz = [](int nPad) { return (size_t)SACT_USH * 2 + (size_t)nPad * 512; };

  // x1 = bn0(conv(x, w1))  -> x1f (f32), x1b (bf16)
  conv_stage<0, 1, 3, 3, 1, 1, 1, 1, 1, 1, 1, 0, -1, -1, 9, 10>
      <<<grid, NTHREADS, shsz(10), stream>>>(
          xb, blob2, bn_gamma, bn_beta, bn_mean, bn_var, x1f, x1b, nullptr, nullptr);
  // s234 = bn1(c(x1,w2)) + bn2(c(x1,w3)) + bn3(c(x1,w4)) -> s234, t1b = relu
  conv_stage<1, 3, 3, 3, 1, 5, 5, 3, 7, 7, 5, 1, 2, 3, 329, 330>
      <<<grid, NTHREADS, shsz(330), stream>>>(
          x1b, blob3, bn_gamma, bn_beta, bn_mean, bn_var, s234, t1b, nullptr, nullptr);
  // xs = relu(x1 + s234 + (bn4..6 convs of t1)) -> xsb (bf16)
  conv_stage<2, 3, 3, 3, 1, 5, 5, 3, 7, 7, 5, 4, 5, 6, 329, 330>
      <<<grid, NTHREADS, shsz(330), stream>>>(
          t1b, blob4, bn_gamma, bn_beta, bn_mean, bn_var, nullptr, xsb, x1f, s234);
  // yacc = bn7(c(x,wr1)) + bn8(c(x,wr2)) + bn9(c(x,wr3)) -> yacc (f32)
  conv_stage<3, 3, 3, 3, 1, 5, 5, 3, 7, 7, 5, 7, 8, 9, 329, 330>
      <<<grid, NTHREADS, shsz(330), stream>>>(
          xb, blob5a, bn_gamma, bn_beta, bn_mean, bn_var, yacc, nullptr, nullptr, nullptr);
  // out = relu(x1 + conv(xs, wch) + yacc) -> d_out (NCDHW f32)
  conv_stage<4, 1, 1, 1, 1, 1, 1, 1, 1, 1, 1, -1, -1, -1, 1, 2>
      <<<grid, NTHREADS, shsz(2), stream>>>(
          xsb, blob5b, bn_gamma, bn_beta, bn_mean, bn_var, out, nullptr, x1f, yacc);

  (void)in_sizes; (void)n_in; (void)out_size; (void)ws_size;
}
